// IndoorVFE_58205396796027
// MI455X (gfx1250) — matrix operand order repaired and checked by reading, not compiled, not run on hardware
//
#include <hip/hip_runtime.h>
#include <hip/hip_bf16.h>

// ---------------------------------------------------------------------------
// IndoorVFE (voxel scatter-mean) for MI455X / gfx1250, wave32.
//
// Pipeline (all state in d_ws, L2-resident on MI455X's 192MB L2):
//   1. Per point: FNV64 hash + 20-bit packed key; atomic count + atomicMin
//      first-occurrence index per key (matches stable argsort semantics).
//   2. Rank present keys by hash order: 22-bit hash-prefix counting sort
//      (4M bins, avg occupancy 0.25) -> exact global rank of each key.
//   3. Feature scatter-add into output row = rank: async-load each wave's
//      32x64 f32 tile to LDS (ASYNCcnt path), then per-wave WMMA one-hot
//      segment combine (v_wmma_f32_16x16x32_f16) with branch-free atomics.
//   4. Normalize rows by counts, emit first-occurrence coords and v2p map.
// ---------------------------------------------------------------------------

#define IVFE_KS (1u << 20)          // key space (32^4)
#define IVFE_BS (1u << 22)          // hash-prefix bins
#define IVFE_BSHIFT 42              // 64 - 22
#define IVFE_NB (IVFE_BS / 256u)    // scan blocks

#if defined(__HIP_DEVICE_COMPILE__) && defined(__gfx1250__) && \
    __has_builtin(__builtin_amdgcn_global_load_async_to_lds_b128)
#define IVFE_ASYNC 1
#else
#define IVFE_ASYNC 0
#endif

typedef __attribute__((ext_vector_type(16))) _Float16 v16h_t;
typedef __attribute__((ext_vector_type(8)))  float    v8f_t;
typedef __attribute__((__vector_size__(16))) int      ivfe_i32x4;

__device__ __forceinline__ void ivfe_wait_async() {
#if IVFE_ASYNC
#if __has_builtin(__builtin_amdgcn_s_wait_asynccnt)
  __builtin_amdgcn_s_wait_asynccnt(0);
#else
  asm volatile("s_wait_asynccnt 0" ::: "memory");
#endif
#endif
}

__device__ __forceinline__ unsigned long long ivfe_fnv4(unsigned c0, unsigned c1,
                                                        unsigned c2, unsigned c3) {
  const unsigned long long P = 1099511628211ull;
  unsigned long long h = 14695981039346656037ull;
  h = (h * P) ^ (unsigned long long)c0;
  h = (h * P) ^ (unsigned long long)c1;
  h = (h * P) ^ (unsigned long long)c2;
  h = (h * P) ^ (unsigned long long)c3;
  return h;
}

// ---------------- generic u32 fill ----------------
__global__ void ivfe_fill_u32(unsigned* p, unsigned val, long long n) {
  long long i = (long long)blockIdx.x * blockDim.x + threadIdx.x;
  if (i < n) p[i] = val;
}

// ---------------- 1: hash, key, count, min-index ----------------
__global__ void ivfe_hash_count(const int* __restrict__ coords,
                                unsigned* __restrict__ keyPt,
                                unsigned* __restrict__ cnt,
                                unsigned* __restrict__ minidx, int N) {
  int i = blockIdx.x * blockDim.x + threadIdx.x;
  if (i >= N) return;
  int4 c = ((const int4*)coords)[i];
  unsigned key = ((unsigned)c.x << 15) | ((unsigned)c.y << 10) |
                 ((unsigned)c.z << 5) | (unsigned)c.w;
  keyPt[i] = key;
  atomicAdd(&cnt[key], 1u);
  atomicMin(&minidx[key], (unsigned)i);
}

// ---------------- 2a: bin histogram over present keys ----------------
__global__ void ivfe_bin_count(const unsigned* __restrict__ cnt,
                               unsigned* __restrict__ binCnt) {
  unsigned k = blockIdx.x * blockDim.x + threadIdx.x;
  if (k >= IVFE_KS) return;
  if (!cnt[k]) return;
  unsigned long long h = ivfe_fnv4((k >> 15) & 31u, (k >> 10) & 31u,
                                   (k >> 5) & 31u, k & 31u);
  atomicAdd(&binCnt[(unsigned)(h >> IVFE_BSHIFT)], 1u);
}

// ---------------- 2b: exclusive scan of 4M bins (3-pass) ----------------
__global__ void ivfe_scan_block(const unsigned* __restrict__ in,
                                unsigned* __restrict__ out,
                                unsigned* __restrict__ bsum) {
  __shared__ unsigned tmp[256];
  int tid = threadIdx.x;
  unsigned gid = blockIdx.x * 256u + tid;
  unsigned v = in[gid];
  tmp[tid] = v;
  __syncthreads();
  for (int off = 1; off < 256; off <<= 1) {
    unsigned t = (tid >= off) ? tmp[tid - off] : 0u;
    __syncthreads();
    tmp[tid] += t;
    __syncthreads();
  }
  out[gid] = tmp[tid] - v;                 // exclusive within block
  if (tid == 255) bsum[blockIdx.x] = tmp[tid];
}

__global__ void ivfe_scan_sums(unsigned* __restrict__ bsum, int nb) {
  __shared__ unsigned tmp[256];
  __shared__ unsigned carry;
  int tid = threadIdx.x;
  if (tid == 0) carry = 0u;
  __syncthreads();
  for (int c0 = 0; c0 < nb; c0 += 256) {
    unsigned v = bsum[c0 + tid];
    tmp[tid] = v;
    __syncthreads();
    for (int off = 1; off < 256; off <<= 1) {
      unsigned t = (tid >= off) ? tmp[tid - off] : 0u;
      __syncthreads();
      tmp[tid] += t;
      __syncthreads();
    }
    unsigned incl = tmp[tid];
    unsigned run = carry;
    __syncthreads();
    bsum[c0 + tid] = run + incl - v;       // exclusive across chunks
    if (tid == 255) carry = run + incl;
    __syncthreads();
  }
}

__global__ void ivfe_scan_add(unsigned* __restrict__ out,
                              const unsigned* __restrict__ bsum) {
  unsigned gid = blockIdx.x * 256u + threadIdx.x;
  out[gid] += bsum[blockIdx.x];
}

__global__ void ivfe_total(const unsigned* __restrict__ binStart,
                           const unsigned* __restrict__ binCnt,
                           unsigned* __restrict__ misc) {
  misc[0] = binStart[IVFE_BS - 1] + binCnt[IVFE_BS - 1];
}

// ---------------- 2c: compact present keys into bins ----------------
__global__ void ivfe_compact(const unsigned* __restrict__ cnt,
                             const unsigned* __restrict__ binStart,
                             unsigned* __restrict__ binFill,
                             unsigned long long* __restrict__ chash,
                             unsigned* __restrict__ ckey) {
  unsigned k = blockIdx.x * blockDim.x + threadIdx.x;
  if (k >= IVFE_KS) return;
  if (!cnt[k]) return;
  unsigned long long h = ivfe_fnv4((k >> 15) & 31u, (k >> 10) & 31u,
                                   (k >> 5) & 31u, k & 31u);
  unsigned b = (unsigned)(h >> IVFE_BSHIFT);
  unsigned pos = binStart[b] + atomicAdd(&binFill[b], 1u);
  chash[pos] = h;
  ckey[pos] = k;
}

// ---------------- 2d: global rank = binStart + within-bin rank ----------------
__global__ void ivfe_rank(const unsigned long long* __restrict__ chash,
                          const unsigned* __restrict__ ckey,
                          const unsigned* __restrict__ binStart,
                          const unsigned* __restrict__ binCnt,
                          const unsigned* __restrict__ misc,
                          unsigned* __restrict__ rank) {
  unsigned p = blockIdx.x * blockDim.x + threadIdx.x;
  if (p >= misc[0]) return;
  unsigned long long h = chash[p];
  unsigned b = (unsigned)(h >> IVFE_BSHIFT);
  unsigned s = binStart[b], n = binCnt[b];
  unsigned rk = s;
  for (unsigned q = 0; q < n; ++q) {       // avg bin size 0.25
    unsigned long long hq = chash[s + q];
    rk += (hq < h) || (hq == h && (s + q) < p);
  }
  rank[ckey[p]] = rk;
}

// ---------------- 3: WMMA segment-combine scatter-add ----------------
// Per wave of 32 points: async-stage the 32x64 f32 tile into LDS, build
// one-hot A(16x32) mapping each point to the slot of the first lane sharing
// its voxel rank, D = A x B accumulates per-slot channel sums in f32 from f16-rounded features.
// Atomics are branch-free: inactive slots are redirected to a dump row.
__global__ void ivfe_accum(const float* __restrict__ feats,
                           const unsigned* __restrict__ keyPt,
                           const unsigned* __restrict__ rank,
                           float* __restrict__ outF,
                           float* __restrict__ dump, int N) {
  int lane = threadIdx.x & 31;
  long long w = ((long long)blockIdx.x * blockDim.x + threadIdx.x) >> 5;
  long long base = w * 32;
  if (base >= N) return;
  bool full = (base + 32 <= N);
  long long i = base + lane;
  unsigned r = 0xFFFFFFFFu;
  if (i < N) r = rank[keyPt[i]];

#if IVFE_ASYNC
  __shared__ float ldsF[8][2048];          // 8 waves x 32 pts x 64 ch = 64 KB
  int wib = (threadIdx.x >> 5) & 7;
#endif

  if (full) {
#if IVFE_ASYNC
    // stage 8KB tile: 16 x GLOBAL_LOAD_ASYNC_TO_LDS_B128 (512B per issue)
    {
      const char* gsrc = (const char*)(feats + base * 64);
      char* lbase = (char*)&ldsF[wib][0];
      int off0 = lane * 16;
      for (int it = 0; it < 16; ++it) {
        int off = off0 + it * 512;
        __builtin_amdgcn_global_load_async_to_lds_b128(
            (__attribute__((address_space(1))) ivfe_i32x4*)(gsrc + off),
            (__attribute__((address_space(3))) ivfe_i32x4*)(lbase + off),
            0, 0);
      }
    }
#else
    __builtin_prefetch(feats + (base + 32) * 64, 0, 1);
#endif

    // lowest lane with same rank (all lanes run all iterations: wave32-safe)
    int first = 0;
    for (int l = 31; l >= 0; --l) {
      unsigned rl = (unsigned)__shfl((int)r, l, 32);
      if (rl == r) first = l;
    }
    unsigned leaders = (unsigned)__ballot(first == lane);
    int slot = __popc(leaders & ((1u << first) - 1u));  // slot of my leader
    int nslots = __popc(leaders);
    int wslots = nslots < 16 ? nslots : 16;

    // ---- build one-hot A (16x32 f16): lane<16 holds row=lane, K in {0..7,16..23};
    //      lane>=16 holds row=lane-16, K in {8..15,24..31}
    int row = lane & 15;
    int kb0 = (lane < 16) ? 0 : 8;
    int kb1 = (lane < 16) ? 16 : 24;
    v16h_t a;
    for (int e = 0; e < 8; ++e) {
      int sj = __shfl(slot, kb0 + e, 32);
      a[e] = (_Float16)((sj == row) ? 1.0f : 0.0f);
    }
    for (int e = 0; e < 8; ++e) {
      int sj = __shfl(slot, kb1 + e, 32);
      a[8 + e] = (_Float16)((sj == row) ? 1.0f : 0.0f);
    }

    // destination row pointer per output row m = mbase+v (dump row if unused);
    // k-th set bit via msk &= msk-1 extraction, then uniform-exec shuffles
    int mbase = (lane < 16) ? 0 : 8;
    unsigned msk = leaders;
    for (int s = 0; s < mbase; ++s) msk &= msk - 1u;
    int lnArr[8];
    for (int v = 0; v < 8; ++v) {
      lnArr[v] = msk ? (__ffs(msk) - 1) : 0;
      msk &= msk - 1u;
    }
    float* dstArr[8];
    for (int v = 0; v < 8; ++v) {
      unsigned rs = (unsigned)__shfl((int)r, lnArr[v], 32);
      dstArr[v] = (mbase + v < wslots) ? (outF + (long long)rs * 64) : dump;
    }

#if IVFE_ASYNC
    ivfe_wait_async();                      // LDS tile ready for this wave
    const float* fsrc = &ldsF[wib][(lane < 16 ? 0 : 8) * 64];
#else
    const float* fsrc = feats + (base + (lane < 16 ? 0 : 8)) * 64;
#endif

    // ---- 4 channel tiles of 16: B is 32(pts) x 16(ch)
    int col = lane & 15;
    for (int t = 0; t < 4; ++t) {
      v16h_t b;
      for (int e = 0; e < 16; ++e)
        b[e] = (_Float16)fsrc[(long long)(e < 8 ? e : e + 8) * 64 + t * 16 + col];
      v8f_t c = {};
      c = __builtin_amdgcn_wmma_f32_16x16x32_f16(
          /*neg_a=*/false, a, /*neg_b=*/false, b,
          /*c_mod=*/(short)0, c, /*reuse_a=*/false, /*reuse_b=*/false);
      for (int v = 0; v < 8; ++v)
        atomicAdd(dstArr[v] + t * 16 + col, c[v]);   // branch-free
    }
    // overflow slots (>= 16 distinct voxels in wave): scalar per-point add
    if (slot >= 16) {
#if IVFE_ASYNC
      const float4* f4 = (const float4*)&ldsF[wib][lane * 64];
#else
      const float4* f4 = (const float4*)(feats + i * 64);
#endif
      float* o = outF + (long long)r * 64;
      for (int q = 0; q < 16; ++q) {
        float4 v = f4[q];
        atomicAdd(&o[q * 4 + 0], v.x);
        atomicAdd(&o[q * 4 + 1], v.y);
        atomicAdd(&o[q * 4 + 2], v.z);
        atomicAdd(&o[q * 4 + 3], v.w);
      }
    }
    return;
  }

  // ragged tail: plain scalar atomics
  if (i < N) {
    const float4* f4 = (const float4*)(feats + i * 64);
    float* o = outF + (long long)r * 64;
    for (int q = 0; q < 16; ++q) {
      float4 v = f4[q];
      atomicAdd(&o[q * 4 + 0], v.x);
      atomicAdd(&o[q * 4 + 1], v.y);
      atomicAdd(&o[q * 4 + 2], v.z);
      atomicAdd(&o[q * 4 + 3], v.w);
    }
  }
}

// ---------------- 4: normalize + voxel coords ----------------
__global__ void ivfe_finalize(const unsigned* __restrict__ cnt,
                              const unsigned* __restrict__ minidx,
                              const unsigned* __restrict__ rank,
                              const int* __restrict__ coords,
                              float* __restrict__ outF,
                              float* __restrict__ outC) {
  unsigned k = blockIdx.x * blockDim.x + threadIdx.x;
  if (k >= IVFE_KS) return;
  unsigned c = cnt[k];
  if (!c) return;
  unsigned r = rank[k];
  float inv = 1.0f / (float)c;
  float4* row = (float4*)(outF + (long long)r * 64);
  for (int q = 0; q < 16; ++q) {
    float4 v = row[q];
    v.x *= inv; v.y *= inv; v.z *= inv; v.w *= inv;
    row[q] = v;
  }
  unsigned mi = minidx[k];
  int4 cc = ((const int4*)coords)[mi];
  outC[(long long)r * 4 + 0] = (float)cc.x;
  outC[(long long)r * 4 + 1] = (float)cc.y;
  outC[(long long)r * 4 + 2] = (float)cc.z;
  outC[(long long)r * 4 + 3] = (float)cc.w;
}

// ---------------- 5: point -> voxel map ----------------
__global__ void ivfe_v2p(const unsigned* __restrict__ keyPt,
                         const unsigned* __restrict__ rank,
                         float* __restrict__ outM, int N) {
  int i = blockIdx.x * blockDim.x + threadIdx.x;
  if (i >= N) return;
  outM[i] = (float)rank[keyPt[i]];
}

// ---------------------------------------------------------------------------
extern "C" void kernel_launch(void* const* d_in, const int* in_sizes, int n_in,
                              void* d_out, int out_size, void* d_ws, size_t ws_size,
                              hipStream_t stream) {
  (void)n_in; (void)ws_size;
  const int* coords = (const int*)d_in[0];
  const float* feats = (const float*)d_in[1];
  const int N = in_sizes[0] / 4;   // coords are [N,4]

  // workspace carve-up (~76 MB; histogram state stays L2-resident)
  char* ws = (char*)d_ws;
  size_t off = 0;
  auto take = [&](size_t bytes) -> char* {
    char* p = ws + off;
    off += (bytes + 255) & ~(size_t)255;
    return p;
  };
  unsigned* keyPt             = (unsigned*)take((size_t)4 * N);
  unsigned* cnt               = (unsigned*)take((size_t)4 * IVFE_KS);
  unsigned* minidx            = (unsigned*)take((size_t)4 * IVFE_KS);
  unsigned* rank              = (unsigned*)take((size_t)4 * IVFE_KS);
  unsigned long long* chash   = (unsigned long long*)take((size_t)8 * IVFE_KS);
  unsigned* ckey              = (unsigned*)take((size_t)4 * IVFE_KS);
  unsigned* binCnt            = (unsigned*)take((size_t)4 * IVFE_BS);
  unsigned* binStart          = (unsigned*)take((size_t)4 * IVFE_BS);
  unsigned* binFill           = (unsigned*)take((size_t)4 * IVFE_BS);
  unsigned* blockSums         = (unsigned*)take((size_t)4 * IVFE_NB);
  unsigned* misc              = (unsigned*)take(256);
  float*    dump              = (float*)take(256);   // branch-free atomic sink

  float* outF = (float*)d_out;                       // [N,64]
  float* outC = outF + (long long)N * 64;            // [N,4]
  float* outM = outC + (long long)N * 4;             // [N]
  (void)out_size;

  const int T = 256;
  const int gN  = (N + T - 1) / T;
  const int gKS = IVFE_KS / T;
  const int gBS = IVFE_BS / T;

  // init
  ivfe_fill_u32<<<gKS, T, 0, stream>>>(cnt, 0u, IVFE_KS);
  ivfe_fill_u32<<<gKS, T, 0, stream>>>(minidx, 0xFFFFFFFFu, IVFE_KS);
  ivfe_fill_u32<<<gBS, T, 0, stream>>>(binCnt, 0u, IVFE_BS);
  ivfe_fill_u32<<<gBS, T, 0, stream>>>(binFill, 0u, IVFE_BS);
  {
    long long nz = (long long)N * 69;  // full output region
    int gz = (int)((nz + T - 1) / T);
    ivfe_fill_u32<<<gz, T, 0, stream>>>((unsigned*)d_out, 0u, nz);
  }

  // 1: hash + histogram
  ivfe_hash_count<<<gN, T, 0, stream>>>(coords, keyPt, cnt, minidx, N);
  // 2: rank present keys by hash order
  ivfe_bin_count<<<gKS, T, 0, stream>>>(cnt, binCnt);
  ivfe_scan_block<<<gBS, T, 0, stream>>>(binCnt, binStart, blockSums);
  ivfe_scan_sums<<<1, T, 0, stream>>>(blockSums, IVFE_NB);
  ivfe_scan_add<<<gBS, T, 0, stream>>>(binStart, blockSums);
  ivfe_total<<<1, 1, 0, stream>>>(binStart, binCnt, misc);
  ivfe_compact<<<gKS, T, 0, stream>>>(cnt, binStart, binFill, chash, ckey);
  ivfe_rank<<<gKS, T, 0, stream>>>(chash, ckey, binStart, binCnt, misc, rank);
  // 3: async-staged WMMA segment-combine feature accumulation
  ivfe_accum<<<gN, T, 0, stream>>>(feats, keyPt, rank, outF, dump, N);
  // 4/5: normalize, coords, map
  ivfe_finalize<<<gKS, T, 0, stream>>>(cnt, minidx, rank, coords, outF, outC);
  ivfe_v2p<<<gN, T, 0, stream>>>(keyPt, rank, outM, N);
}
